// VADLite_661424964312
// MI455X (gfx1250) — matrix operand order repaired and checked by reading, not compiled, not run on hardware
//
#include <hip/hip_runtime.h>
#include <hip/hip_bf16.h>

// ---------------- problem constants ----------------
#define NB 8
#define NA 64
#define NLANES 256      // L
#define NPTS 20         // P
#define SD 8
#define MF 6
#define HD 256          // H
#define NHEADS 8
#define HDIM 32         // H / NH
#define NL 3
#define NC 64           // C
#define NS 30           // S
#define NTOK 321        // 1 + A + L
#define NTOK_PAD 336    // 21*16
#define ROWS (NB*NTOK_PAD)   // 2688
#define KDIM 256
#define TRAJN 3840      // C*S*2
#define DT_ 0.1f
#define MAX_SPEED_ 15.0f
#define MAX_ACCEL_ 5.0f
#define MAX_CURV_ 0.3f
#define SAFETY_ 1.5f
#define MAX_STEP_ 1.5f

// ---------------- workspace layout ----------------
#define H_TOK    ((size_t)0)                   // 2688*256
#define H_WK     (H_TOK + (size_t)ROWS*KDIM)   // 256*256
#define H_WV     (H_WK + 65536)
#define H_WTRAJ  (H_WV + 65536)                // 3840*256
#define H_FUSED  (H_WTRAJ + (size_t)TRAJN*KDIM) // 16*256
#define HALF_BYTES ((H_FUSED + 4096) * 2)      // 3,612,672 bytes
#define F_KMAT   ((size_t)0)                   // 2688*256
#define F_VMAT   (F_KMAT + (size_t)ROWS*KDIM)
#define F_FUSED  (F_VMAT + (size_t)ROWS*KDIM)  // 8*256
#define F_TRAJRAW (F_FUSED + 2048)             // 16*3840
#define F_LDIR   (F_TRAJRAW + (size_t)16*TRAJN)// 16
#define F_VALID  (F_LDIR + 16)                 // 512 (int view)

// ---------------- output layout (floats) ----------------
#define O_TRAJ    0
#define O_LOGITS  30720
#define O_SCORES  31232
#define O_SEL     31744
#define O_SELTRAJ 31752
#define O_MIND    32232
#define O_LANED   32744
#define O_ALIGN   33256
#define O_COST    33768

typedef __attribute__((ext_vector_type(16))) _Float16 v16h;
typedef __attribute__((ext_vector_type(8)))  _Float16 v8h;
typedef __attribute__((ext_vector_type(8)))  float    v8f;

// ---------------- helpers ----------------
__device__ __forceinline__ float warpSum32(float v) {
#pragma unroll
  for (int o = 16; o > 0; o >>= 1) v += __shfl_xor(v, o, 32);
  return v;
}
__device__ __forceinline__ float warpMax32(float v) {
#pragma unroll
  for (int o = 16; o > 0; o >>= 1) v = fmaxf(v, __shfl_xor(v, o, 32));
  return v;
}
__device__ __forceinline__ float blockSum256(float v, float* s8) {
  int lane = threadIdx.x & 31, w = threadIdx.x >> 5;
  v = warpSum32(v);
  __syncthreads();
  if (lane == 0) s8[w] = v;
  __syncthreads();
  float t = 0.f;
#pragma unroll
  for (int i = 0; i < 8; i++) t += s8[i];
  return t;
}

// ---------------- 1) weight conversion to f16 ----------------
__global__ void k_convw(const float* __restrict__ attn_in_w,
                        const float* __restrict__ traj_w,
                        _Float16* __restrict__ wk, _Float16* __restrict__ wv,
                        _Float16* __restrict__ wtraj) {
  int i = blockIdx.x * blockDim.x + threadIdx.x;
  if (i < 65536) {
    wk[i] = (_Float16)attn_in_w[65536 + i];   // rows H..2H  (W_k)
    wv[i] = (_Float16)attn_in_w[131072 + i];  // rows 2H..3H (W_v)
  }
  if (i < TRAJN * KDIM) {
    int n = i / KDIM, k = i - n * KDIM;       // transpose (K,N)->(N,K)
    wtraj[(size_t)n * KDIM + k] = (_Float16)traj_w[(size_t)k * TRAJN + n];
  }
}

// ---------------- 2) token encoder + LN ----------------
__global__ void k_encode(const float* __restrict__ ego, const float* __restrict__ agents,
                         const float* __restrict__ mp,
                         const float* __restrict__ wE, const float* __restrict__ bE,
                         const float* __restrict__ wA, const float* __restrict__ bA,
                         const float* __restrict__ wM, const float* __restrict__ bM,
                         const float* __restrict__ lng, const float* __restrict__ lnb,
                         _Float16* __restrict__ tokf16, float* __restrict__ fused) {
  int blk = blockIdx.x;
  int b = blk / NTOK, n = blk % NTOK;
  int t = threadIdx.x;
  __shared__ float f[8];
  __shared__ float s8[8];
  const float* W; const float* bias; int nd;
  if (n == 0)        { W = wE; bias = bE; nd = 8; }
  else if (n <= NA)  { W = wA; bias = bA; nd = 8; }
  else               { W = wM; bias = bM; nd = 6; }
  if (t < nd) {
    if (n == 0)       f[t] = ego[(b * 10 + 9) * SD + t];
    else if (n <= NA) f[t] = agents[((size_t)b * NA + (n - 1)) * SD + t];
    else {
      float s = 0.f;
      size_t base = (((size_t)b * NLANES + (n - 1 - NA)) * NPTS) * MF + t;
#pragma unroll
      for (int p = 0; p < NPTS; p++) s += mp[base + (size_t)p * MF];
      f[t] = s * (1.0f / NPTS);
    }
  }
  __syncthreads();
  float val = bias[t];
  for (int d = 0; d < nd; d++) val += f[d] * W[d * HD + t];
  float mean = blockSum256(val, s8) * (1.0f / HD);
  float dv = val - mean;
  float var = blockSum256(dv * dv, s8) * (1.0f / HD);
  float outv = dv * rsqrtf(var + 1e-5f) * lng[t] + lnb[t];
  tokf16[((size_t)b * NTOK_PAD + n) * KDIM + t] = (_Float16)outv;
  if (n == 0) fused[(size_t)b * HD + t] = outv;
}

// ---------------- 3a) WMMA GEMM with async-LDS weight staging ----------------
// Block = 4 waves; all share one 16xK weight tile staged global->LDS via
// global_load_async_to_lds_b128 (ASYNCcnt path), each wave owns one M-tile.
__global__ void __launch_bounds__(128)
k_gemm16_lds(const _Float16* __restrict__ A, const _Float16* __restrict__ Wt,
             const float* __restrict__ bias, float* __restrict__ C, int K, int ldc) {
  __shared__ alignas(64) _Float16 sW[16 * KDIM];   // 8 KB weight tile
  int t = threadIdx.x;
  int wave = t >> 5, lane = t & 31;
  int lm = lane & 15, hf = lane >> 4;
  int nt = blockIdx.y;
  int mt = blockIdx.x * 4 + wave;

  // --- async copy: 16 rows x K halves = 16*K*2 bytes, 16B chunks over 128 lanes ---
  {
    const char* wtile = (const char*)(Wt + (size_t)(nt * 16) * K);
    unsigned lds_base = (unsigned)(uintptr_t)(&sW[0]);   // LDS aperture: low 32 bits = LDS offset
    int nbytes = 16 * K * 2;
#pragma unroll
    for (int it = 0; it < 4; ++it) {
      int off = (it * 128 + t) * 16;
      if (off < nbytes) {
        unsigned lds = lds_base + (unsigned)off;
        unsigned long long ga = (unsigned long long)(uintptr_t)(wtile + off);
        asm volatile("global_load_async_to_lds_b128 %0, %1, off"
                     :: "v"(lds), "v"(ga) : "memory");
      }
    }
    asm volatile("s_wait_asynccnt 0x0" ::: "memory");
  }
  __syncthreads();

  const _Float16* arow = A + (size_t)(mt * 16 + lm) * K;
  const _Float16* wrow = sW + (size_t)lm * K;
  v8f acc = {};
  for (int kb = 0; kb < K; kb += 32) {
    union { v16h v; v8h p[2]; } ua;
    ua.p[0] = *(const v8h*)(arow + kb + 16 * hf);       // K = 16h..16h+7
    ua.p[1] = *(const v8h*)(arow + kb + 16 * hf + 8);   // K = 16h+8..16h+15
    v16h bf = *(const v16h*)(wrow + kb + 16 * hf);      // ds_load from staged tile
    acc = __builtin_amdgcn_wmma_f32_16x16x32_f16(false, ua.v, false, bf,
                                                 (short)0, acc, false, false);
  }
  int col = nt * 16 + lm;
  float bv = bias ? bias[col] : 0.f;
#pragma unroll
  for (int r = 0; r < 8; r++) {
    int row = mt * 16 + r + 8 * hf;
    C[(size_t)row * ldc + col] = acc[r] + bv;
  }
}

// ---------------- 3b) single-wave WMMA GEMM (small M, direct from global) ----------------
__global__ void __launch_bounds__(32)
k_gemm16(const _Float16* __restrict__ A, const _Float16* __restrict__ Wt,
         const float* __restrict__ bias, float* __restrict__ C, int K, int ldc) {
  int mt = blockIdx.x, nt = blockIdx.y;
  int lane = threadIdx.x;
  int lm = lane & 15, hf = lane >> 4;
  const _Float16* arow = A + (size_t)(mt * 16 + lm) * K;
  const _Float16* wrow = Wt + (size_t)(nt * 16 + lm) * K;
  v8f acc = {};
  for (int kb = 0; kb < K; kb += 32) {
    union { v16h v; v8h p[2]; } ua;
    ua.p[0] = *(const v8h*)(arow + kb + 16 * hf);
    ua.p[1] = *(const v8h*)(arow + kb + 16 * hf + 8);
    v16h bf = *(const v16h*)(wrow + kb + 16 * hf);
    acc = __builtin_amdgcn_wmma_f32_16x16x32_f16(false, ua.v, false, bf,
                                                 (short)0, acc, false, false);
  }
  int col = nt * 16 + lm;
  float bv = bias ? bias[col] : 0.f;
#pragma unroll
  for (int r = 0; r < 8; r++) {
    int row = mt * 16 + r + 8 * hf;
    C[(size_t)row * ldc + col] = acc[r] + bv;
  }
}

// ---------------- 4) fused 3-layer attention + MLP + LN (one block per batch) ----------------
__global__ void __launch_bounds__(256)
k_attn(const float* __restrict__ Kmat, const float* __restrict__ Vmat,
       const float* __restrict__ attn_in_w, const float* __restrict__ attn_in_b,
       const float* __restrict__ attn_out_w, const float* __restrict__ attn_out_b,
       const float* __restrict__ mlp_w1, const float* __restrict__ mlp_b1,
       const float* __restrict__ mlp_w2, const float* __restrict__ mlp_b2,
       const float* __restrict__ pl_g, const float* __restrict__ pl_b,
       float* __restrict__ fused_io, _Float16* __restrict__ fused_f16) {
  int b = blockIdx.x, t = threadIdx.x;
  int head = t >> 5, lane = t & 31;
  __shared__ float sf[HD];
  __shared__ float sq[HD];
  __shared__ float satt[NHEADS * NTOK_PAD];
  __shared__ float sattn[HD];
  __shared__ float sh1[2 * HD];
  __shared__ float shs[NHEADS];
  __shared__ float s8[8];
  sf[t] = fused_io[(size_t)b * HD + t];
  __syncthreads();
  const float* Kb = Kmat + (size_t)b * NTOK_PAD * KDIM;
  const float* Vb = Vmat + (size_t)b * NTOK_PAD * KDIM;
  for (int i = 0; i < NL; i++) {
    {
      const float* wq = attn_in_w + (size_t)t * HD;
      float a = attn_in_b[t];
      for (int d = 0; d < HD; d++) a += sf[d] * wq[d];
      sq[t] = a;
    }
    __syncthreads();
    float mx = -1e30f;
    for (int n = lane; n < NTOK; n += 32) {
      const float* kr = Kb + (size_t)n * KDIM + head * HDIM;
      float s = 0.f;
#pragma unroll
      for (int d = 0; d < HDIM; d++) s += sq[head * HDIM + d] * kr[d];
      s *= 0.17677669529663687f;   // 1/sqrt(32)
      satt[head * NTOK_PAD + n] = s;
      mx = fmaxf(mx, s);
    }
    mx = warpMax32(mx);
    float sm = 0.f;
    for (int n = lane; n < NTOK; n += 32) {
      float e = __expf(satt[head * NTOK_PAD + n] - mx);
      satt[head * NTOK_PAD + n] = e;
      sm += e;
    }
    sm = warpSum32(sm);
    if (lane == 0) shs[head] = 1.0f / sm;
    __syncthreads();
    float acc = 0.f;
    for (int n = 0; n < NTOK; n++)
      acc += satt[head * NTOK_PAD + n] * Vb[(size_t)n * KDIM + t];
    sattn[t] = acc * shs[head];
    __syncthreads();
    float ao = attn_out_b[t];
    {
      const float* wo = attn_out_w + (size_t)t * HD;
      for (int d = 0; d < HD; d++) ao += sattn[d] * wo[d];
    }
    for (int j = t; j < 2 * HD; j += HD) {
      float p = mlp_b1[i * 2 * HD + j];
      const float* w1 = mlp_w1 + ((size_t)i * HD) * (2 * HD) + j;
      for (int d = 0; d < HD; d++) p += sf[d] * w1[(size_t)d * (2 * HD)];
      sh1[j] = 0.5f * p * (1.0f + erff(p * 0.70710678118654752f)); // exact gelu
    }
    __syncthreads();
    float m2 = mlp_b2[i * HD + t];
    {
      const float* w2 = mlp_w2 + ((size_t)i * 2 * HD) * HD + t;
      for (int j = 0; j < 2 * HD; j++) m2 += sh1[j] * w2[(size_t)j * HD];
    }
    float res = sf[t] + ao + m2;
    float mean = blockSum256(res, s8) * (1.0f / HD);
    float dv = res - mean;
    float var = blockSum256(dv * dv, s8) * (1.0f / HD);
    float nf = dv * rsqrtf(var + 1e-5f) * pl_g[t] + pl_b[t];
    __syncthreads();
    sf[t] = nf;
    __syncthreads();
  }
  fused_io[(size_t)b * HD + t] = sf[t];
  fused_f16[(size_t)b * KDIM + t] = (_Float16)sf[t];
}

// ---------------- 5) score head + softmax ----------------
__global__ void k_score(const float* __restrict__ fused, const float* __restrict__ score_w,
                        const float* __restrict__ score_b, float* __restrict__ out) {
  int b = blockIdx.x, c = threadIdx.x;
  float l = score_b[c];
  for (int d = 0; d < HD; d++) l += fused[(size_t)b * HD + d] * score_w[d * NC + c];
  __shared__ float sl[NC];
  __shared__ float se[NC];
  sl[c] = l; __syncthreads();
  float mx = -1e30f;
  for (int j = 0; j < NC; j++) mx = fmaxf(mx, sl[j]);
  float e = __expf(l - mx);
  se[c] = e; __syncthreads();
  float s = 0.f;
  for (int j = 0; j < NC; j++) s += se[j];
  out[O_LOGITS + b * NC + c] = l;
  out[O_SCORES + b * NC + c] = e / s;
}

// ---------------- 6) trajectory build: tanh + cumsum ----------------
__global__ void k_trajbuild(const float* __restrict__ trajraw, const float* __restrict__ ego,
                            float* __restrict__ out) {
  int g = blockIdx.x * blockDim.x + threadIdx.x;
  if (g >= NB * NC) return;
  int b = g >> 6, c = g & 63;
  float x = ego[(b * 10 + 9) * SD + 0];
  float y = ego[(b * 10 + 9) * SD + 1];
  const float* raw = trajraw + (size_t)b * TRAJN + c * (NS * 2);
  float* to = out + O_TRAJ + (size_t)(b * NC + c) * (NS * 2);
  for (int s = 0; s < NS; s++) {
    x += tanhf(raw[s * 2 + 0]) * MAX_STEP_;
    y += tanhf(raw[s * 2 + 1]) * MAX_STEP_;
    to[s * 2 + 0] = x; to[s * 2 + 1] = y;
  }
}

// ---------------- 7) mean lane direction per batch ----------------
__global__ void k_ldir(const float* __restrict__ mp, float* __restrict__ ldir) {
  int b = blockIdx.x, l = threadIdx.x;
  __shared__ float s8[8];
  size_t base = ((size_t)b * NLANES + l) * NPTS * MF;
  float dx = mp[base + (NPTS - 1) * MF + 0] - mp[base + 0];
  float dy = mp[base + (NPTS - 1) * MF + 1] - mp[base + 1];
  float sx = blockSum256(dx, s8) * (1.0f / NLANES);
  float sy = blockSum256(dy, s8) * (1.0f / NLANES);
  if (l == 0) {
    float n = fmaxf(sqrtf(sx * sx + sy * sy), 1e-6f);
    ldir[b * 2 + 0] = sx / n; ldir[b * 2 + 1] = sy / n;
  }
}

// ---------------- 8) per-(b,c) geometry metrics ----------------
__global__ void __launch_bounds__(256)
k_metrics(const float* __restrict__ out_traj, const float* __restrict__ agents,
          const float* __restrict__ mp, const float* __restrict__ ldir,
          float* __restrict__ out, int* __restrict__ vflag) {
  int b = blockIdx.x >> 6, c = blockIdx.x & 63, t = threadIdx.x;
  __shared__ float tx[NS], ty[NS];
  __shared__ float r1[256], r2[256];
  if (t < NS) {
    size_t base = ((size_t)(b * NC + c) * NS + t) * 2;
    tx[t] = out_traj[O_TRAJ + base];
    ty[t] = out_traj[O_TRAJ + base + 1];
  }
  __syncthreads();
  float la = 1e30f;
  for (int idx = t; idx < NS * NA; idx += 256) {
    int s = idx >> 6, a = idx & 63;
    float ax = agents[((size_t)b * NA + a) * SD + 0];
    float ay = agents[((size_t)b * NA + a) * SD + 1];
    float dx = tx[s] - ax, dy = ty[s] - ay;
    la = fminf(la, dx * dx + dy * dy);
  }
  float ll = 1e30f;
  for (int j = t; j < NLANES * NPTS; j += 256) {
    float lx = mp[((size_t)b * NLANES * NPTS + j) * MF + 0];
    float ly = mp[((size_t)b * NLANES * NPTS + j) * MF + 1];
#pragma unroll 6
    for (int s = 0; s < NS; s++) {
      float dx = tx[s] - lx, dy = ty[s] - ly;
      ll = fminf(ll, dx * dx + dy * dy);
    }
  }
  r1[t] = la; r2[t] = ll;
  __syncthreads();
  for (int st = 128; st > 0; st >>= 1) {
    if (t < st) { r1[t] = fminf(r1[t], r1[t + st]); r2[t] = fminf(r2[t], r2[t + st]); }
    __syncthreads();
  }
  if (t == 0) {
    float mind = sqrtf(r1[0]);
    float laned = sqrtf(r2[0]);
    float maxv = 0.f, maxa = 0.f, maxc = 0.f;
    for (int s = 0; s < NS - 1; s++) {
      float dx = tx[s + 1] - tx[s], dy = ty[s + 1] - ty[s];
      maxv = fmaxf(maxv, sqrtf(dx * dx + dy * dy) / DT_);
    }
    for (int s = 0; s < NS - 2; s++) {
      float d1x = tx[s + 1] - tx[s],   d1y = ty[s + 1] - ty[s];
      float d2x = tx[s + 2] - tx[s + 1], d2y = ty[s + 2] - ty[s + 1];
      float ddx = d2x - d1x, ddy = d2y - d1y;
      maxa = fmaxf(maxa, sqrtf(ddx * ddx + ddy * ddy) / (DT_ * DT_));
      float cross = d1x * ddy - d1y * ddx;
      float l1 = sqrtf(d1x * d1x + d1y * d1y);
      maxc = fmaxf(maxc, fabsf(cross) / fmaxf(l1 * l1 * l1, 1e-6f));
    }
    int feasible = (maxv <= MAX_SPEED_) && (maxa <= MAX_ACCEL_) && (maxc <= MAX_CURV_);
    float tdx = tx[NS - 1] - tx[0], tdy = ty[NS - 1] - ty[0];
    float tn = fmaxf(sqrtf(tdx * tdx + tdy * tdy), 1e-6f);
    float align = (tdx * ldir[b * 2 + 0] + tdy * ldir[b * 2 + 1]) / tn;
    int valid = feasible && (mind >= SAFETY_) && (laned <= SAFETY_ * 3.0f) && (align >= 0.0f);
    float cost = 1.0f * fmaxf(SAFETY_ - mind, 0.f) + 0.1f * laned + 0.5f * fmaxf(1.0f - align, 0.f);
    int o = b * NC + c;
    out[O_MIND + o] = mind;
    out[O_LANED + o] = laned;
    out[O_ALIGN + o] = align;
    out[O_COST + o] = cost;
    vflag[o] = valid;
  }
}

// ---------------- 9) selection ----------------
__global__ void k_select(const float* __restrict__ outr, const int* __restrict__ vflag,
                         float* __restrict__ out) {
  int b = blockIdx.x, t = threadIdx.x;
  __shared__ int ssel;
  if (t == 0) {
    int any = 0, is = 0, ir = 0;
    float bs = -1e30f, br = -1e30f;
    for (int c = 0; c < NC; c++) {
      float sc = outr[O_SCORES + b * NC + c];
      int v = vflag[b * NC + c];
      any |= v;
      float sv = v ? sc : -1.0f;
      if (sv > bs) { bs = sv; is = c; }
      if (sc > br) { br = sc; ir = c; }
    }
    int sel = any ? is : ir;
    out[O_SEL + b] = (float)sel;
    ssel = sel;
  }
  __syncthreads();
  if (t < NS * 2)
    out[O_SELTRAJ + b * (NS * 2) + t] =
        outr[O_TRAJ + (size_t)(b * NC + ssel) * (NS * 2) + t];
}

// ---------------- launch ----------------
extern "C" void kernel_launch(void* const* d_in, const int* in_sizes, int n_in,
                              void* d_out, int out_size, void* d_ws, size_t ws_size,
                              hipStream_t stream) {
  const float* ego        = (const float*)d_in[0];
  const float* agents     = (const float*)d_in[1];
  const float* mp         = (const float*)d_in[2];
  const float* enc_ego_w  = (const float*)d_in[3];
  const float* enc_ego_b  = (const float*)d_in[4];
  const float* enc_agent_w= (const float*)d_in[5];
  const float* enc_agent_b= (const float*)d_in[6];
  const float* enc_map_w  = (const float*)d_in[7];
  const float* enc_map_b  = (const float*)d_in[8];
  const float* enc_ln_g   = (const float*)d_in[9];
  const float* enc_ln_b   = (const float*)d_in[10];
  const float* attn_in_w  = (const float*)d_in[11];
  const float* attn_in_b  = (const float*)d_in[12];
  const float* attn_out_w = (const float*)d_in[13];
  const float* attn_out_b = (const float*)d_in[14];
  const float* mlp_w1     = (const float*)d_in[15];
  const float* mlp_b1     = (const float*)d_in[16];
  const float* mlp_w2     = (const float*)d_in[17];
  const float* mlp_b2     = (const float*)d_in[18];
  const float* pl_ln_g    = (const float*)d_in[19];
  const float* pl_ln_b    = (const float*)d_in[20];
  const float* traj_w     = (const float*)d_in[21];
  const float* traj_b     = (const float*)d_in[22];
  const float* score_w    = (const float*)d_in[23];
  const float* score_b    = (const float*)d_in[24];

  _Float16* wsh = (_Float16*)d_ws;
  float*    wsf = (float*)((char*)d_ws + HALF_BYTES);
  float*    out = (float*)d_out;

  _Float16* tokf16 = wsh + H_TOK;
  _Float16* wk16   = wsh + H_WK;
  _Float16* wv16   = wsh + H_WV;
  _Float16* wt16   = wsh + H_WTRAJ;
  _Float16* fu16   = wsh + H_FUSED;
  float* Kmat   = wsf + F_KMAT;
  float* Vmat   = wsf + F_VMAT;
  float* fused  = wsf + F_FUSED;
  float* trajraw= wsf + F_TRAJRAW;
  float* ldir   = wsf + F_LDIR;
  int*   vflag  = (int*)(wsf + F_VALID);

  // 1) f16 weight staging
  k_convw<<<(TRAJN * KDIM + 255) / 256, 256, 0, stream>>>(attn_in_w, traj_w, wk16, wv16, wt16);
  // 2) encode tokens + LN
  k_encode<<<NB * NTOK, 256, 0, stream>>>(ego, agents, mp, enc_ego_w, enc_ego_b,
                                          enc_agent_w, enc_agent_b, enc_map_w, enc_map_b,
                                          enc_ln_g, enc_ln_b, tokf16, fused);
  // 3) WMMA K/V projections with async-LDS weight staging: (2688 x 256) @ (256 x 256)
  {
    dim3 g(ROWS / 64, KDIM / 16);   // 4 M-tiles per block (4 waves)
    k_gemm16_lds<<<g, 128, 0, stream>>>(tokf16, wk16, attn_in_b + HD,     Kmat, KDIM, KDIM);
    k_gemm16_lds<<<g, 128, 0, stream>>>(tokf16, wv16, attn_in_b + 2 * HD, Vmat, KDIM, KDIM);
  }
  // 4) fused attention stack
  k_attn<<<NB, 256, 0, stream>>>(Kmat, Vmat, attn_in_w, attn_in_b, attn_out_w, attn_out_b,
                                 mlp_w1, mlp_b1, mlp_w2, mlp_b2, pl_ln_g, pl_ln_b,
                                 fused, fu16);
  // 5) WMMA trajectory head: (16 x 256) @ (256 x 3840), single M-tile
  {
    dim3 g(1, TRAJN / 16);
    k_gemm16<<<g, 32, 0, stream>>>(fu16, wt16, traj_b, trajraw, KDIM, TRAJN);
  }
  // 6) scores
  k_score<<<NB, NC, 0, stream>>>(fused, score_w, score_b, out);
  // 7) traj build
  k_trajbuild<<<(NB * NC + 255) / 256, 256, 0, stream>>>(trajraw, ego, out);
  // 8) lane direction
  k_ldir<<<NB, 256, 0, stream>>>(mp, ldir);
  // 9) metrics per (b,c)
  k_metrics<<<NB * NC, 256, 0, stream>>>(out, agents, mp, ldir, out, vflag);
  // 10) selection
  k_select<<<NB, 64, 0, stream>>>(out, vflag, out);
}
